// MySeq2SeqAttention_83683142795679
// MI455X (gfx1250) — matrix operand order repaired and checked by reading, not compiled, not run on hardware
//
#include <hip/hip_runtime.h>
#include <hip/hip_bf16.h>
#include <math.h>

// ---------------------------------------------------------------------------
// Seq2seq w/ attention on gfx1250. All matmuls on v_wmma_f32_16x16x32_bf16.
// Weights + activations materialized in bf16 once; WMMA inner loops are pure
// bf16 b128 loads + wmma. LSTM step kernel stages its shared A-tile in LDS
// via the gfx1250 async-to-LDS path (ASYNCcnt) when available.
// B=S=T=128, SRC=TRG=512, ENC_H=512, DEC_H=1024.
// ---------------------------------------------------------------------------

typedef __attribute__((ext_vector_type(16))) __bf16 v16bf;
typedef __attribute__((ext_vector_type(8)))  __bf16 v8bf;
typedef __attribute__((ext_vector_type(4)))  __bf16 v4bf;
typedef __attribute__((ext_vector_type(8)))  float  v8f;
typedef __attribute__((ext_vector_type(4)))  int    v4i;

#define NB 128
#define NS 128
#define NT 128

#if __has_builtin(__builtin_amdgcn_global_load_async_to_lds_b128) && \
    __has_builtin(__builtin_amdgcn_s_wait_asynccnt)
#define HAS_ASYNC_LDS 1
#else
#define HAS_ASYNC_LDS 0
#endif

__device__ __forceinline__ float sigmoidf_(float x) { return 1.f / (1.f + __expf(-x)); }

// A fragment 16x32 (MxK), bf16: p = row base + half*8 (+k0).
// elems 0..7 = 8 contiguous bf16 at p, elems 8..15 = 8 contiguous at p+16.
__device__ __forceinline__ v16bf load_a_frag(const __bf16* p) {
  v8bf x0 = *(const v8bf*)(p);
  v8bf x1 = *(const v8bf*)(p + 16);
  return __builtin_shufflevector(x0, x1, 0, 1, 2, 3, 4, 5, 6, 7,
                                 8, 9, 10, 11, 12, 13, 14, 15);
}

// B fragment 32x16 (KxN) fed from W rows (C = A @ W^T): p = W row n + half*8 (+k0);
// elems 0..7 = 8 contiguous bf16 at p, elems 8..15 = 8 contiguous at p+16.
__device__ __forceinline__ v16bf load_b_frag(const __bf16* p) {
  v8bf x0 = *(const v8bf*)(p);
  v8bf x1 = *(const v8bf*)(p + 16);
  return __builtin_shufflevector(x0, x1, 0, 1, 2, 3, 4, 5, 6, 7,
                                 8, 9, 10, 11, 12, 13, 14, 15);
}

__device__ __forceinline__ v8f wmma_bf16(v16bf a, v16bf b, v8f c) {
  return __builtin_amdgcn_wmma_f32_16x16x32_bf16(false, a, false, b, (short)0, c, false, false);
}

// ---------------------------------------------------------------------------
// f32 -> bf16 conversion pass (4 elements / thread, n multiple of 4)
// ---------------------------------------------------------------------------
__global__ void cvt_bf16_kernel(const float* __restrict__ in, __bf16* __restrict__ out, int n) {
  int i4 = (blockIdx.x * blockDim.x + threadIdx.x) * 4;
  if (i4 >= n) return;
  float4 v = *(const float4*)(in + i4);
  v4bf o;
  o[0] = (__bf16)v.x; o[1] = (__bf16)v.y; o[2] = (__bf16)v.z; o[3] = (__bf16)v.w;
  *(v4bf*)(out + i4) = o;
}

// ---------------------------------------------------------------------------
// Generic GEMM: act(A[M,K](bf16) @ W[N,K](bf16)^T + bias(f32)).
// Output: Cbf (bf16) if non-null, else Cf (f32). One wave = one 16x16 tile.
// ---------------------------------------------------------------------------
__global__ void gemm_kernel(const __bf16* __restrict__ A, int lda,
                            const __bf16* __restrict__ W, int ldw,
                            const float* __restrict__ bias,
                            float* __restrict__ Cf, __bf16* __restrict__ Cbf, int ldc,
                            int M, int N, int K, int act) {
  int wave = (int)((blockIdx.x * blockDim.x + threadIdx.x) >> 5);
  int lane = threadIdx.x & 31;
  int tilesN = N >> 4;
  int tm = wave / tilesN;
  int tn = wave % tilesN;
  if (tm * 16 >= M) return;  // uniform per wave
  int half = (lane >> 4) & 1;
  int l16 = lane & 15;

  const __bf16* arow = A + (size_t)(tm * 16 + l16) * lda + half * 8;
  const __bf16* brow = W + (size_t)(tn * 16 + l16) * ldw + half * 8;

  v8f acc = {};
  for (int k0 = 0; k0 < K; k0 += 32) {
    __builtin_prefetch(arow + k0 + 128, 0, 1);  // global_prefetch_b8
    acc = wmma_bf16(load_a_frag(arow + k0), load_b_frag(brow + k0), acc);
  }

  int n = tn * 16 + l16;
  float bv = bias ? bias[n] : 0.f;
#pragma unroll
  for (int r = 0; r < 8; ++r) {
    int m = tm * 16 + half * 8 + r;
    float v = acc[r] + bv;
    if (act) v = tanhf(v);
    if (Cbf) Cbf[(size_t)m * ldc + n] = (__bf16)v;
    else     Cf[(size_t)m * ldc + n] = v;
  }
}

// ---------------------------------------------------------------------------
// Stage a 16 x dim bf16 tile (row stride ld) into LDS (row pitch = dim+24,
// which staggers the 16 lanes' 16B reads across disjoint bank groups).
// Uses gfx1250 async-to-LDS when available; 128 threads, dim in {512,1024}.
// ---------------------------------------------------------------------------
__device__ __forceinline__ void stage_tile(const __bf16* __restrict__ g, int ld,
                                           int dim, __bf16* lds, int tid) {
  int cpr = dim >> 3;          // 16B chunks per row
  int chunks = 16 * cpr;       // total chunks (multiple of 128)
  int pitch = dim + 24;
  for (int ci = tid; ci < chunks; ci += 128) {
    int row = ci / cpr;
    int c8 = (ci % cpr) * 8;
    const __bf16* gp = g + (size_t)row * ld + c8;
    __bf16* lp = lds + row * pitch + c8;
#if HAS_ASYNC_LDS
    __builtin_amdgcn_global_load_async_to_lds_b128(
        (__attribute__((address_space(1))) v4i*)gp,
        (__attribute__((address_space(3))) v4i*)lp, 0, 0);
#else
    *(v8bf*)lp = *(const v8bf*)gp;
#endif
  }
#if HAS_ASYNC_LDS
  __builtin_amdgcn_s_wait_asynccnt(0);
#endif
  __syncthreads();
}

// ---------------------------------------------------------------------------
// Fused LSTM step (torch gate order i,f,g,o):
// gates = x@Wih^T + h@Whh^T + bih + bhh; c' = sig(f)*c + sig(i)*tanh(g);
// h' = sig(o)*tanh(c'). Block = 4 waves sharing one 16-row M tile; the A tile
// is staged once per phase in LDS (async path) and read with ds_load_b128,
// while each wave streams its own 4 gate-weight B tiles from L2.
// h' always written bf16 (feeds the next WMMA); optional extra f32 copy.
// first=1 => h_prev = c_prev = 0.
// ---------------------------------------------------------------------------
__global__ void lstm_step_kernel(const __bf16* __restrict__ x, int ldx, int in_dim,
                                 const __bf16* __restrict__ h_in, int ldh,
                                 const __bf16* __restrict__ Wih,
                                 const __bf16* __restrict__ Whh,
                                 const float* __restrict__ bih,
                                 const float* __restrict__ bhh,
                                 float* __restrict__ c_st, int ldcst,
                                 __bf16* __restrict__ h_bf, int ldho,
                                 float* __restrict__ h_f32,
                                 int H, int first) {
  __shared__ __bf16 As[16 * 1048];  // 16 x (<=1024 + 24 pad) bf16 = ~33KB

  int tid = threadIdx.x;
  int lane = tid & 31;
  int tilesJ4 = (H >> 4) >> 2;             // tj groups of 4 per M tile
  int tm = blockIdx.x / tilesJ4;
  int tj = (blockIdx.x % tilesJ4) * 4 + (tid >> 5);
  int half = (lane >> 4) & 1;
  int l16 = lane & 15;

  v8f acc_i = {}, acc_f = {}, acc_g = {}, acc_o = {};

  {  // ---- x-side: K = in_dim, A from LDS ----
    stage_tile(x + (size_t)tm * 16 * ldx, ldx, in_dim, As, tid);
    int pitch = in_dim + 24;
    const __bf16* arow = As + l16 * pitch + half * 8;
    const __bf16* b0 = Wih + (size_t)(0 * H + tj * 16 + l16) * in_dim + half * 8;
    const __bf16* b1 = Wih + (size_t)(1 * H + tj * 16 + l16) * in_dim + half * 8;
    const __bf16* b2 = Wih + (size_t)(2 * H + tj * 16 + l16) * in_dim + half * 8;
    const __bf16* b3 = Wih + (size_t)(3 * H + tj * 16 + l16) * in_dim + half * 8;
    for (int k0 = 0; k0 < in_dim; k0 += 32) {
      v16bf a = load_a_frag(arow + k0);
      acc_i = wmma_bf16(a, load_b_frag(b0 + k0), acc_i);
      acc_f = wmma_bf16(a, load_b_frag(b1 + k0), acc_f);
      acc_g = wmma_bf16(a, load_b_frag(b2 + k0), acc_g);
      acc_o = wmma_bf16(a, load_b_frag(b3 + k0), acc_o);
    }
    __syncthreads();  // done reading As before h-phase restage
  }
  if (!first) {  // ---- h-side: K = H, A from LDS ----
    stage_tile(h_in + (size_t)tm * 16 * ldh, ldh, H, As, tid);
    int pitch = H + 24;
    const __bf16* arow = As + l16 * pitch + half * 8;
    const __bf16* b0 = Whh + (size_t)(0 * H + tj * 16 + l16) * H + half * 8;
    const __bf16* b1 = Whh + (size_t)(1 * H + tj * 16 + l16) * H + half * 8;
    const __bf16* b2 = Whh + (size_t)(2 * H + tj * 16 + l16) * H + half * 8;
    const __bf16* b3 = Whh + (size_t)(3 * H + tj * 16 + l16) * H + half * 8;
    for (int k0 = 0; k0 < H; k0 += 32) {
      v16bf a = load_a_frag(arow + k0);
      acc_i = wmma_bf16(a, load_b_frag(b0 + k0), acc_i);
      acc_f = wmma_bf16(a, load_b_frag(b1 + k0), acc_f);
      acc_g = wmma_bf16(a, load_b_frag(b2 + k0), acc_g);
      acc_o = wmma_bf16(a, load_b_frag(b3 + k0), acc_o);
    }
  }

  int j = tj * 16 + l16;
  float bi = bih[0 * H + j] + bhh[0 * H + j];
  float bf = bih[1 * H + j] + bhh[1 * H + j];
  float bg = bih[2 * H + j] + bhh[2 * H + j];
  float bo = bih[3 * H + j] + bhh[3 * H + j];
#pragma unroll
  for (int r = 0; r < 8; ++r) {
    int m = tm * 16 + half * 8 + r;
    float gi = sigmoidf_(acc_i[r] + bi);
    float gf = sigmoidf_(acc_f[r] + bf);
    float gg = tanhf(acc_g[r] + bg);
    float go = sigmoidf_(acc_o[r] + bo);
    float cp = first ? 0.f : c_st[(size_t)m * ldcst + j];
    float cn = gf * cp + gi * gg;
    c_st[(size_t)m * ldcst + j] = cn;
    float hv = go * tanhf(cn);
    h_bf[(size_t)m * ldho + j] = (__bf16)hv;
    if (h_f32) h_f32[(size_t)m * ldho + j] = hv;
  }
}

// ---------------------------------------------------------------------------
// Soft-dot attention for one decoder step. One block per batch row b.
// scores[s] = ctx[b,s,:]·target[b,:]; softmax; wctx -> cat_bf[b, 0:1024].
// ctx f32 [B,S,1024]; cat_bf row stride 2048 (hy lives in cols 1024+).
// ---------------------------------------------------------------------------
__global__ void attn_step_kernel(const float* __restrict__ ctx,
                                 const float* __restrict__ tgt,
                                 __bf16* __restrict__ cat_bf) {
  __shared__ float sc[NS];
  __shared__ float red[NS];
  int b = blockIdx.x;
  int s = threadIdx.x;

  const float* crow = ctx + (size_t)b * NS * 1024 + (size_t)s * 1024;
  const float* trow = tgt + (size_t)b * 1024;
  float acc = 0.f;
  for (int d = 0; d < 1024; d += 4) {
    float4 cv = *(const float4*)(crow + d);
    float4 tv = *(const float4*)(trow + d);
    acc += cv.x * tv.x + cv.y * tv.y + cv.z * tv.z + cv.w * tv.w;
  }
  sc[s] = acc;
  red[s] = acc;
  __syncthreads();
  for (int off = 64; off; off >>= 1) {
    if (s < off) red[s] = fmaxf(red[s], red[s + off]);
    __syncthreads();
  }
  float mx = red[0];
  __syncthreads();
  float e = __expf(sc[s] - mx);
  sc[s] = e;
  red[s] = e;
  __syncthreads();
  for (int off = 64; off; off >>= 1) {
    if (s < off) red[s] += red[s + off];
    __syncthreads();
  }
  float inv = 1.f / red[0];
  __syncthreads();
  sc[s] = sc[s] * inv;
  __syncthreads();

  const float* cbase = ctx + (size_t)b * NS * 1024;
#pragma unroll
  for (int i = 0; i < 8; ++i) {
    int d = s + i * 128;
    float a2 = 0.f;
    for (int s2 = 0; s2 < NS; ++s2) a2 += sc[s2] * cbase[(size_t)s2 * 1024 + d];
    cat_bf[(size_t)b * 2048 + d] = (__bf16)a2;
  }
}

// hcat[b,0:512]=src_h_bf[b,0,512:1024] (h_bwd); hcat[b,512:]=src_h_bf[b,S-1,0:512]
__global__ void concat_h_kernel(const __bf16* __restrict__ src_h_bf, __bf16* __restrict__ hcat) {
  int idx = blockIdx.x * blockDim.x + threadIdx.x;  // B*1024
  int b = idx >> 10;
  int k = idx & 1023;
  __bf16 v = (k < 512) ? src_h_bf[(size_t)b * NS * 1024 + 512 + k]
                       : src_h_bf[(size_t)b * NS * 1024 + (size_t)(NS - 1) * 1024 + (k - 512)];
  hcat[idx] = v;
}

// ---------------------------------------------------------------------------
// Host-side orchestration
// ---------------------------------------------------------------------------
static inline void launch_cvt(hipStream_t st, const float* in, __bf16* out, int n) {
  cvt_bf16_kernel<<<(n / 4 + 255) / 256, 256, 0, st>>>(in, out, n);
}

static inline void launch_gemm(hipStream_t st, const __bf16* A, int lda,
                               const __bf16* W, int ldw, const float* bias,
                               float* Cf, __bf16* Cbf, int ldc,
                               int M, int N, int K, int act) {
  int waves = (M / 16) * (N / 16);
  gemm_kernel<<<(waves + 3) / 4, 128, 0, st>>>(A, lda, W, ldw, bias, Cf, Cbf, ldc, M, N, K, act);
}

static inline void launch_lstm(hipStream_t st, const __bf16* x, int ldx, int in_dim,
                               const __bf16* h_in, int ldh,
                               const __bf16* Wih, const __bf16* Whh,
                               const float* bih, const float* bhh,
                               float* c, int ldcst, __bf16* h_bf, int ldho, float* h_f32,
                               int H, int first) {
  int blocks = (NB / 16) * ((H / 16) / 4);  // 4 waves/block share one M tile
  lstm_step_kernel<<<blocks, 128, 0, st>>>(x, ldx, in_dim, h_in, ldh, Wih, Whh, bih, bhh,
                                           c, ldcst, h_bf, ldho, h_f32, H, first);
}

extern "C" void kernel_launch(void* const* d_in, const int* in_sizes, int n_in,
                              void* d_out, int out_size, void* d_ws, size_t ws_size,
                              hipStream_t stream) {
  const float* F[32];
  for (int i = 0; i < 32; ++i) F[i] = (const float*)d_in[i];
  const float* b_src = F[3];
  const float* b_trg = F[5];
  const float* ebih[4] = {F[8], F[12], F[16], F[20]};
  const float* ebhh[4] = {F[9], F[13], F[17], F[21]};
  const float* b_e2d = F[23];
  const float* dec_bi = F[25];
  const float* dec_bh = F[27];
  const float* b_d2o = F[31];

  // ---- workspace bump allocator (256B aligned) ----
  char* p = (char*)d_ws;
  auto alloc = [&](size_t bytes) -> char* {
    char* r = p;
    p += (bytes + 255) & ~(size_t)255;
    return r;
  };
  auto abf = [&](size_t elems) -> __bf16* { return (__bf16*)alloc(elems * 2); };
  auto af32 = [&](size_t elems) -> float* { return (float*)alloc(elems * 4); };

  // bf16 weight copies (converted once; L2-resident thereafter)
  __bf16* W_src_bf = abf(512 * 512);
  __bf16* W_trg_bf = abf(512 * 512);
  __bf16* eWih_bf[4];
  __bf16* eWhh_bf[4];
  const int wih_rows[4] = {2048, 2048, 2048, 2048};
  const int wih_cols[4] = {512, 512, 1024, 1024};
  for (int l = 0; l < 4; ++l) {
    eWih_bf[l] = abf((size_t)wih_rows[l] * wih_cols[l]);
    eWhh_bf[l] = abf(2048 * 512);
  }
  __bf16* W_e2d_bf = abf(1024 * 1024);
  __bf16* dec_Wi_bf = abf(4096 * 512);
  __bf16* dec_Wh_bf = abf((size_t)4096 * 1024);
  __bf16* attn_Win_bf = abf(1024 * 1024);
  __bf16* attn_Wout_bf = abf((size_t)1024 * 2048);
  __bf16* W_d2o_bf = abf(512 * 1024);

  // bf16 activations
  __bf16* in_src_bf = abf((size_t)NB * NS * 512);
  __bf16* in_trg_bf = abf((size_t)NB * NT * 512);
  __bf16* src_emb = abf((size_t)NB * NS * 512);   // [B,S,512]
  __bf16* trg_emb = abf((size_t)NB * NT * 512);   // [B,T,512]
  __bf16* x1 = abf((size_t)NB * NS * 1024);       // layer0 out [fwd|bwd]
  __bf16* src_h_bf = abf((size_t)NB * NS * 1024); // layer1 out (WMMA feed)
  __bf16* trg_h = abf((size_t)NB * NT * 1024);    // decoder h_tilde chain
  __bf16* hcat = abf((size_t)NB * 1024);          // [h_bwd|h_fwd]
  __bf16* h0 = abf((size_t)NB * 1024);            // decoder init h
  __bf16* cat = abf((size_t)NB * 2048);           // [wctx | hy]

  // f32 buffers
  float* src_h = af32((size_t)NB * NS * 1024);    // ctx for attention (VALU)
  float* c0f = af32((size_t)NB * 512);
  float* c0b = af32((size_t)NB * 512);
  float* ccat = af32((size_t)NB * 1024);          // [c_bwd|c_fwd] -> decoder c
  float* tgt = af32((size_t)NB * 1024);           // attn target

  // ---- 0) one-time f32 -> bf16 conversions ----
  launch_cvt(stream, F[0], in_src_bf, NB * NS * 512);
  launch_cvt(stream, F[1], in_trg_bf, NB * NT * 512);
  launch_cvt(stream, F[2], W_src_bf, 512 * 512);
  launch_cvt(stream, F[4], W_trg_bf, 512 * 512);
  const int wih_idx[4] = {6, 10, 14, 18};
  for (int l = 0; l < 4; ++l) {
    launch_cvt(stream, F[wih_idx[l]], eWih_bf[l], wih_rows[l] * wih_cols[l]);
    launch_cvt(stream, F[wih_idx[l] + 1], eWhh_bf[l], 2048 * 512);
  }
  launch_cvt(stream, F[22], W_e2d_bf, 1024 * 1024);
  launch_cvt(stream, F[24], dec_Wi_bf, 4096 * 512);
  launch_cvt(stream, F[26], dec_Wh_bf, 4096 * 1024);
  launch_cvt(stream, F[28], attn_Win_bf, 1024 * 1024);
  launch_cvt(stream, F[29], attn_Wout_bf, 1024 * 2048);
  launch_cvt(stream, F[30], W_d2o_bf, 512 * 1024);

  // ---- 1) embeddings (batched WMMA GEMMs, bf16 out) ----
  launch_gemm(stream, in_src_bf, 512, W_src_bf, 512, b_src,
              nullptr, src_emb, 512, NB * NS, 512, 512, 0);
  launch_gemm(stream, in_trg_bf, 512, W_trg_bf, 512, b_trg,
              nullptr, trg_emb, 512, NB * NT, 512, 512, 0);

  // ---- 2) encoder layer 0 (h chains through x1 itself) ----
  for (int t = 0; t < NS; ++t) {
    launch_lstm(stream, src_emb + (size_t)t * 512, NS * 512, 512,
                t ? x1 + (size_t)(t - 1) * 1024 : nullptr, NS * 1024,
                eWih_bf[0], eWhh_bf[0], ebih[0], ebhh[0],
                c0f, 512, x1 + (size_t)t * 1024, NS * 1024, nullptr, 512, t == 0);
  }
  for (int tt = 0; tt < NS; ++tt) {
    int t = NS - 1 - tt;
    launch_lstm(stream, src_emb + (size_t)t * 512, NS * 512, 512,
                tt ? x1 + (size_t)(t + 1) * 1024 + 512 : nullptr, NS * 1024,
                eWih_bf[1], eWhh_bf[1], ebih[1], ebhh[1],
                c0b, 512, x1 + (size_t)t * 1024 + 512, NS * 1024, nullptr, 512, tt == 0);
  }

  // ---- 3) encoder layer 1 (dual bf16+f32 h out; final c lands in ccat) ----
  for (int t = 0; t < NS; ++t) {
    launch_lstm(stream, x1 + (size_t)t * 1024, NS * 1024, 1024,
                t ? src_h_bf + (size_t)(t - 1) * 1024 : nullptr, NS * 1024,
                eWih_bf[2], eWhh_bf[2], ebih[2], ebhh[2],
                ccat + 512, 1024, src_h_bf + (size_t)t * 1024, NS * 1024,
                src_h + (size_t)t * 1024, 512, t == 0);
  }
  for (int tt = 0; tt < NS; ++tt) {
    int t = NS - 1 - tt;
    launch_lstm(stream, x1 + (size_t)t * 1024, NS * 1024, 1024,
                tt ? src_h_bf + (size_t)(t + 1) * 1024 + 512 : nullptr, NS * 1024,
                eWih_bf[3], eWhh_bf[3], ebih[3], ebhh[3],
                ccat, 1024, src_h_bf + (size_t)t * 1024 + 512, NS * 1024,
                src_h + (size_t)t * 1024 + 512, 512, tt == 0);
  }

  // ---- 4) h0 = tanh([h_bwd|h_fwd] @ W_e2d^T + b) ----
  concat_h_kernel<<<(NB * 1024) / 256, 256, 0, stream>>>(src_h_bf, hcat);
  launch_gemm(stream, hcat, 1024, W_e2d_bf, 1024, b_e2d,
              nullptr, h0, 1024, NB, 1024, 1024, 1);

  // ---- 5) decoder: lstm -> target gemm -> attention -> h_tilde gemm ----
  for (int t = 0; t < NT; ++t) {
    const __bf16* h_in = t ? trg_h + (size_t)(t - 1) * 1024 : h0;
    int ldh = t ? NT * 1024 : 1024;
    launch_lstm(stream, trg_emb + (size_t)t * 512, NT * 512, 512, h_in, ldh,
                dec_Wi_bf, dec_Wh_bf, dec_bi, dec_bh,
                ccat, 1024, cat + 1024, 2048, nullptr, 1024, 0);  // hy -> cat[:,1024:]
    launch_gemm(stream, cat + 1024, 2048, attn_Win_bf, 1024, nullptr,
                tgt, nullptr, 1024, NB, 1024, 1024, 0);           // target (f32)
    attn_step_kernel<<<NB, 128, 0, stream>>>(src_h, tgt, cat);    // wctx -> cat[:,0:1024]
    launch_gemm(stream, cat, 2048, attn_Wout_bf, 2048, nullptr,
                nullptr, trg_h + (size_t)t * 1024, NT * 1024, NB, 1024, 2048, 1);
  }

  // ---- 6) logits = trg_h @ W_d2o^T + b (f32 out to d_out) ----
  launch_gemm(stream, trg_h, 1024, W_d2o_bf, 1024, b_d2o,
              (float*)d_out, nullptr, 512, NB * NT, 512, 1024, 0);
}
